// GeneratorModelDecoder_EMOPIA_35931696399017
// MI455X (gfx1250) — matrix operand order repaired and checked by reading, not compiled, not run on hardware
//
#include <hip/hip_runtime.h>
#include <stdint.h>
#include <stddef.h>

// ---------------- model dimensions (compile-time) ----------------
constexpr int kB  = 32;
constexpr int kT  = 256;
constexpr int kV  = 512;
constexpr int kD  = 576;
constexpr int kH  = 6;
constexpr int kL  = 8;
constexpr int kHD = 96;    // head dim
constexpr int kDE = 384;   // token-embedding dim
constexpr int kFF = 2304;
constexpr int kBT = kB * kT;   // 8192 rows
constexpr float kEPS = 1e-5f;

// Async-to-LDS staging only in the gfx1250 device pass; host pass and other
// targets get a plain direct-global fallback (same kernel signature).
#if defined(__HIP_DEVICE_COMPILE__) && defined(__gfx1250__)
#define GEMM_ASYNC_LDS 1
#else
#define GEMM_ASYNC_LDS 0
#endif

// ---------------- WMMA types ----------------
typedef __attribute__((ext_vector_type(16))) __bf16 v16bf;
typedef __attribute__((ext_vector_type(8)))  float  v8f;
typedef __attribute__((ext_vector_type(4)))  unsigned int v4u;

union Frag { v16bf v; v4u q[2]; };

__device__ __forceinline__ uint16_t f2bf(float f) {
  uint32_t u = __float_as_uint(f);
  u += 0x7fffu + ((u >> 16) & 1u);   // round-to-nearest-even
  return (uint16_t)(u >> 16);
}
__device__ __forceinline__ float bf2f(uint16_t u) {
  return __uint_as_float(((uint32_t)u) << 16);
}

__device__ __forceinline__ v8f wmma_bf16(const Frag& a, const Frag& b, v8f c) {
  // D = A(16x32 bf16) * B(32x16 bf16) + C(f32)
  return __builtin_amdgcn_wmma_f32_16x16x32_bf16(false, a.v, false, b.v,
                                                 (short)0, c, false, false);
}

// A-operand fragment: row-major matrix, row = m0 + (lane&15),
// per-lane K layout: 16 contiguous elems starting at k0+16h (16h folded into p).
__device__ __forceinline__ void load_fragA(Frag& f, const uint16_t* p, int k0) {
  f.q[0] = *(const v4u*)(p + k0);
  f.q[1] = *(const v4u*)(p + k0 + 8);
}
// B-operand fragment: row-major "weight" matrix W (N x K), row = n0 + (lane&15),
// per-lane K layout: 16 contiguous elems starting at k0+16h (16h folded into p).
__device__ __forceinline__ void load_fragB(Frag& f, const uint16_t* p, int k0) {
  f.q[0] = *(const v4u*)(p + k0);
  f.q[1] = *(const v4u*)(p + k0 + 8);
}

// ---------------- fp32 -> bf16 conversion ----------------
__global__ __launch_bounds__(256)
void f32_to_bf16_kernel(const float* __restrict__ s, uint16_t* __restrict__ d, int n) {
  int i = blockIdx.x * 256 + threadIdx.x;
  if (i < n) d[i] = f2bf(s[i]);
}

// ---------------- embedding + condition + positional ----------------
__global__ __launch_bounds__(256)
void embed_kernel(const int* __restrict__ idx, const float* __restrict__ cond,
                  const float* __restrict__ tok, const float* __restrict__ w_cond,
                  const float* __restrict__ b_cond, const float* __restrict__ pos,
                  float* __restrict__ x) {
  const int i  = blockIdx.x * 256 + threadIdx.x;   // over kBT*kD, exact fit
  const int d  = i % kD;
  const int bt = i / kD;
  const int t  = bt % kT;
  const int b  = bt / kT;
  float v;
  if (d < kDE) {
    v = tok[(size_t)idx[bt] * kDE + d] * 19.595917942265423f;  // sqrt(384)
  } else {
    const int dc = d - kDE;
    v = b_cond[dc];
#pragma unroll
    for (int j = 0; j < 4; ++j) v += cond[b * 4 + j] * w_cond[dc * 4 + j];
  }
  x[i] = v + pos[t * kD + d];
}

// ---------------- LayerNorm (fp32 in, bf16 out), one wave per row ----------------
__global__ __launch_bounds__(256)
void ln_bf16_kernel(const float* __restrict__ xin, const float* __restrict__ g,
                    const float* __restrict__ be, uint16_t* __restrict__ ybf) {
  const int lane = threadIdx.x & 31;
  const int wave = threadIdx.x >> 5;
  const size_t row = (size_t)blockIdx.x * 8 + wave;   // row < kBT
  const float* xr = xin + row * kD;
  float s = 0.f, s2 = 0.f;
  for (int j = lane; j < kD; j += 32) { float t = xr[j]; s += t; s2 += t * t; }
#pragma unroll
  for (int o = 16; o; o >>= 1) { s += __shfl_xor(s, o, 32); s2 += __shfl_xor(s2, o, 32); }
  const float mu  = s  * (1.0f / kD);
  const float var = s2 * (1.0f / kD) - mu * mu;
  const float rs  = rsqrtf(var + kEPS);
  uint16_t* yr = ybf + row * kD;
  for (int j = lane; j < kD; j += 32)
    yr[j] = f2bf((xr[j] - mu) * rs * g[j] + be[j]);
}

// ---------------- generic WMMA GEMM: C = A(MxK) * W(NxK)^T ----------------
// Block: 256 threads = 8 waves; block tile 256(M) x 64(N); wave tile 32x64.
// Weight tile (64 rows x 64 k, 8KB) is double-buffer staged into LDS with
// GLOBAL_LOAD_ASYNC_TO_LDS_B128 (ASYNCcnt) and consumed via ds_load_b128.
// OUTMODE: 0 = fp32 out (row-major MxN), 1 = bf16 out, 2 = bf16 V^T scatter (B,H,HD,T)
template<int OUTMODE, bool RELU, bool RESID, bool BIAS>
__global__ __launch_bounds__(256)
void gemm_wmma_kernel(const uint16_t* __restrict__ A, const uint16_t* __restrict__ W,
                      const float* __restrict__ bias, const float* resid,
                      void* out, int M, int N, int K) {
  const int lane = threadIdx.x & 31;
  const int wave = threadIdx.x >> 5;
  const int r = lane & 15, h = lane >> 4;
  const int m0  = blockIdx.x * 256 + wave * 32;   // 8 waves cover 256 rows
  const int n0b = blockIdx.y * 64;
  (void)M;

  const uint16_t* a0p = A + (size_t)(m0 + r) * K + 16 * h;
  const uint16_t* a1p = a0p + (size_t)16 * K;

  const v8f vzero = {};
  v8f acc[2][4];
#pragma unroll
  for (int i = 0; i < 2; ++i)
#pragma unroll
    for (int j = 0; j < 4; ++j) acc[i][j] = vzero;

#if GEMM_ASYNC_LDS
  __shared__ uint16_t lb[2][64 * 64];             // 2 x 8KB double buffer
  const int trow = threadIdx.x >> 2;              // 0..63 : weight row within tile
  const int tseg = (threadIdx.x & 3) << 4;        // 0/16/32/48 : element offset
  const uint16_t* gsrc = W + (size_t)(n0b + trow) * K + tseg;
  const int nch = K >> 6;                         // K / 64 chunks

  auto issue = [&](int c, int buf) {
    const uint64_t g = (uint64_t)(uintptr_t)(gsrc + (c << 6));
    const uint32_t l = (uint32_t)(uintptr_t)&lb[buf][(trow << 6) + tseg];
    // one instruction offset applies to both global src and LDS dst (ISA 15.18.3)
    asm volatile("global_load_async_to_lds_b128 %0, %1, off"
                 :: "v"(l), "v"(g) : "memory");
    asm volatile("global_load_async_to_lds_b128 %0, %1, off offset:16"
                 :: "v"(l), "v"(g) : "memory");
  };

  issue(0, 0);
  for (int c = 0; c < nch; ++c) {
    if (c + 1 < nch) {
      issue(c + 1, (c + 1) & 1);
      asm volatile("s_wait_asynccnt 0x2" ::: "memory");  // stage c landed, c+1 in flight
    } else {
      asm volatile("s_wait_asynccnt 0x0" ::: "memory");
    }
    __syncthreads();                                     // stage c visible to all waves
    const uint16_t* lw = &lb[c & 1][0];
#pragma unroll
    for (int kk = 0; kk < 2; ++kk) {
      const int k0 = kk << 5;
      Frag a0, a1;
      load_fragA(a0, a0p, (c << 6) + k0);
      load_fragA(a1, a1p, (c << 6) + k0);
#pragma unroll
      for (int j = 0; j < 4; ++j) {
        Frag b;
        const uint16_t* bp = lw + (((j << 4) + r) << 6) + (h << 4) + k0;
        b.q[0] = *(const v4u*)(bp);
        b.q[1] = *(const v4u*)(bp + 8);
        acc[0][j] = wmma_bf16(a0, b, acc[0][j]);
        acc[1][j] = wmma_bf16(a1, b, acc[1][j]);
      }
    }
    __syncthreads();                                     // done reading buf before re-fill
  }
#else
  // fallback: direct global fragment loads (host pass / non-gfx1250)
  const uint16_t* bp0 = W + (size_t)(n0b + r) * K + 16 * h;
  for (int k0 = 0; k0 < K; k0 += 32) {
    Frag a0, a1;
    load_fragA(a0, a0p, k0);
    load_fragA(a1, a1p, k0);
#pragma unroll
    for (int j = 0; j < 4; ++j) {
      Frag b;
      load_fragB(b, bp0 + (size_t)(j * 16) * K, k0);
      acc[0][j] = wmma_bf16(a0, b, acc[0][j]);
      acc[1][j] = wmma_bf16(a1, b, acc[1][j]);
    }
  }
#endif

  auto store_tile = [&](v8f c, int mb, int nb) {
    const int col = nb + r;
    float bv = 0.0f;
    if (BIAS) bv = bias[col];
#pragma unroll
    for (int i = 0; i < 8; ++i) {
      const int row = mb + h * 8 + i;       // C layout: VGPR i -> M = i (+8 for hi lanes)
      float v = c[i] + bv;
      if (RESID) v += resid[(size_t)row * N + col];
      if (RELU)  v = fmaxf(v, 0.0f);
      if (OUTMODE == 0) {
        ((float*)out)[(size_t)row * N + col] = v;
      } else if (OUTMODE == 1) {
        ((uint16_t*)out)[(size_t)row * N + col] = f2bf(v);
      } else {
        // scatter v into V^T layout (B,H,HD,T): row=b*T+t, col=hh*HD+d
        const int b  = row >> 8;            // / kT
        const int t  = row & (kT - 1);
        const int hh = col / kHD;
        const int d  = col % kHD;
        ((uint16_t*)out)[(((size_t)(b * kH + hh)) * kHD + d) * kT + t] = f2bf(v);
      }
    }
  };
#pragma unroll
  for (int mi = 0; mi < 2; ++mi)
#pragma unroll
    for (int j = 0; j < 4; ++j)
      store_tile(acc[mi][j], m0 + mi * 16, n0b + j * 16);
}

// ---------------- attention scores: S = Q K^T * scale, causal mask, bf16 out ----------------
__global__ __launch_bounds__(256)
void attn_scores_kernel(const uint16_t* __restrict__ qb, const uint16_t* __restrict__ kb,
                        uint16_t* __restrict__ sb) {
  const int lane = threadIdx.x & 31;
  const int wave = threadIdx.x >> 5;
  const int r = lane & 15, h = lane >> 4;
  const int bh = blockIdx.y;                 // b*H + hh
  const int b  = bh / kH;
  const int hh = bh % kH;
  const int m0 = wave * 32;                  // 8 waves cover T=256 rows
  const int n0 = blockIdx.x * 32;

  const uint16_t* Ab = qb + ((size_t)b * kT) * kD + hh * kHD;
  const uint16_t* Bb = kb + ((size_t)b * kT) * kD + hh * kHD;
  const uint16_t* a0p = Ab + (size_t)(m0 + r) * kD + 16 * h;
  const uint16_t* a1p = a0p + (size_t)16 * kD;
  const uint16_t* b0p = Bb + (size_t)(n0 + r) * kD + 16 * h;
  const uint16_t* b1p = b0p + (size_t)16 * kD;

  v8f c00 = {}, c01 = {}, c10 = {}, c11 = {};
#pragma unroll
  for (int k0 = 0; k0 < kHD; k0 += 32) {
    Frag a0, a1, b0, b1;
    load_fragA(a0, a0p, k0);
    load_fragA(a1, a1p, k0);
    load_fragB(b0, b0p, k0);
    load_fragB(b1, b1p, k0);
    c00 = wmma_bf16(a0, b0, c00);
    c01 = wmma_bf16(a0, b1, c01);
    c10 = wmma_bf16(a1, b0, c10);
    c11 = wmma_bf16(a1, b1, c11);
  }

  uint16_t* srow = sb + (size_t)bh * kT * kT;
  const float sc = 0.10206207261596575f;     // 96^-0.5
  auto st = [&](v8f c, int mb, int nb) {
    const int col = nb + r;
#pragma unroll
    for (int i = 0; i < 8; ++i) {
      const int row = mb + h * 8 + i;
      srow[(size_t)row * kT + col] = (col <= row) ? f2bf(c[i] * sc) : (uint16_t)0xFF80u; // -inf
    }
  };
  st(c00, m0,      n0);
  st(c01, m0,      n0 + 16);
  st(c10, m0 + 16, n0);
  st(c11, m0 + 16, n0 + 16);
}

// ---------------- row softmax over T=256, in-place on bf16, one wave per row ----------------
__global__ __launch_bounds__(256)
void softmax_kernel(uint16_t* sb) {
  const int lane = threadIdx.x & 31;
  const int wave = threadIdx.x >> 5;
  const size_t row = (size_t)blockIdx.x * 8 + wave;   // rows = B*H*T
  uint16_t* pr = sb + row * kT;
  union { v4u q; uint16_t u16[8]; } buf;
  buf.q = *(v4u*)(pr + lane * 8);
  float v[8];
  float mx = -3.0e38f;
#pragma unroll
  for (int j = 0; j < 8; ++j) { v[j] = bf2f(buf.u16[j]); mx = fmaxf(mx, v[j]); }
#pragma unroll
  for (int o = 16; o; o >>= 1) mx = fmaxf(mx, __shfl_xor(mx, o, 32));
  float s = 0.f;
#pragma unroll
  for (int j = 0; j < 8; ++j) { v[j] = __expf(v[j] - mx); s += v[j]; }
#pragma unroll
  for (int o = 16; o; o >>= 1) s += __shfl_xor(s, o, 32);
  const float inv = 1.0f / s;
#pragma unroll
  for (int j = 0; j < 8; ++j) buf.u16[j] = f2bf(v[j] * inv);
  *(v4u*)(pr + lane * 8) = buf.q;
}

// ---------------- attention output: att = P(TxT) * V(TxHD), V stored as V^T(HD,T) ----------------
__global__ __launch_bounds__(256)
void attn_out_kernel(const uint16_t* __restrict__ pb, const uint16_t* __restrict__ vT,
                     uint16_t* __restrict__ attb) {
  const int lane = threadIdx.x & 31;
  const int wave = threadIdx.x >> 5;
  const int r = lane & 15, h = lane >> 4;
  const int bh = blockIdx.y;
  const int b  = bh / kH;
  const int hh = bh % kH;
  const int m0 = wave * 32;
  const int n0 = blockIdx.x * 32;            // 3 tiles over HD=96

  const uint16_t* Ab = pb + (size_t)bh * kT * kT;        // P, lda = T
  const uint16_t* Bb = vT + (size_t)bh * kHD * kT;       // V^T rows = head-dim, ld = T
  const uint16_t* a0p = Ab + (size_t)(m0 + r) * kT + 16 * h;
  const uint16_t* a1p = a0p + (size_t)16 * kT;
  const uint16_t* b0p = Bb + (size_t)(n0 + r) * kT + 16 * h;
  const uint16_t* b1p = b0p + (size_t)16 * kT;

  v8f c00 = {}, c01 = {}, c10 = {}, c11 = {};
  for (int k0 = 0; k0 < kT; k0 += 32) {
    Frag a0, a1, b0, b1;
    load_fragA(a0, a0p, k0);
    load_fragA(a1, a1p, k0);
    load_fragB(b0, b0p, k0);
    load_fragB(b1, b1p, k0);
    c00 = wmma_bf16(a0, b0, c00);
    c01 = wmma_bf16(a0, b1, c01);
    c10 = wmma_bf16(a1, b0, c10);
    c11 = wmma_bf16(a1, b1, c11);
  }

  auto st = [&](v8f c, int mb, int nb) {
    const int col = nb + r;
#pragma unroll
    for (int i = 0; i < 8; ++i) {
      const int row = mb + h * 8 + i;
      attb[((size_t)(b * kT + row)) * kD + hh * kHD + col] = f2bf(c[i]);
    }
  };
  st(c00, m0,      n0);
  st(c01, m0,      n0 + 16);
  st(c10, m0 + 16, n0);
  st(c11, m0 + 16, n0 + 16);
}

// =====================================================================
extern "C" void kernel_launch(void* const* d_in, const int* in_sizes, int n_in,
                              void* d_out, int out_size, void* d_ws, size_t ws_size,
                              hipStream_t stream) {
  (void)in_sizes; (void)n_in; (void)out_size; (void)ws_size;

  const int*   idx       = (const int*)  d_in[0];
  const float* condition = (const float*)d_in[1];
  const float* token_emb = (const float*)d_in[2];
  const float* w_cond    = (const float*)d_in[3];
  const float* b_cond    = (const float*)d_in[4];
  const float* wq        = (const float*)d_in[5];
  const float* wk        = (const float*)d_in[6];
  const float* wv        = (const float*)d_in[7];
  const float* w_proj    = (const float*)d_in[8];
  const float* b_proj    = (const float*)d_in[9];
  const float* ln1_g     = (const float*)d_in[10];
  const float* ln1_b     = (const float*)d_in[11];
  const float* ln2_g     = (const float*)d_in[12];
  const float* ln2_b     = (const float*)d_in[13];
  const float* w_ff1     = (const float*)d_in[14];
  const float* b_ff1     = (const float*)d_in[15];
  const float* w_ff2     = (const float*)d_in[16];
  const float* b_ff2     = (const float*)d_in[17];
  const float* lnf_g     = (const float*)d_in[18];
  const float* lnf_b     = (const float*)d_in[19];
  const float* w_head    = (const float*)d_in[20];
  const float* b_head    = (const float*)d_in[21];
  const float* pos_emb   = (const float*)d_in[22];

  // ---- workspace arena ----
  char* ws = (char*)d_ws;
  size_t off = 0;
  auto alloc = [&](size_t bytes) -> void* {
    off = (off + 255) & ~(size_t)255;
    void* p = ws + off;
    off += bytes;
    return p;
  };
  uint16_t* wq_b  = (uint16_t*)alloc((size_t)kL * kD * kD * 2);
  uint16_t* wk_b  = (uint16_t*)alloc((size_t)kL * kD * kD * 2);
  uint16_t* wv_b  = (uint16_t*)alloc((size_t)kL * kD * kD * 2);
  uint16_t* wp_b  = (uint16_t*)alloc((size_t)kL * kD * kD * 2);
  uint16_t* wf1_b = (uint16_t*)alloc((size_t)kL * kFF * kD * 2);
  uint16_t* wf2_b = (uint16_t*)alloc((size_t)kL * kD * kFF * 2);
  uint16_t* wh_b  = (uint16_t*)alloc((size_t)kV * kD * 2);
  float*    x     = (float*)   alloc((size_t)kBT * kD * 4);
  uint16_t* ybf   = (uint16_t*)alloc((size_t)kBT * kD * 2);
  uint16_t* attbf = (uint16_t*)alloc((size_t)kBT * kD * 2);
  uint16_t* qbf   = (uint16_t*)alloc((size_t)kBT * kD * 2);
  uint16_t* kbf   = (uint16_t*)alloc((size_t)kBT * kD * 2);
  uint16_t* vTbf  = (uint16_t*)alloc((size_t)kBT * kD * 2);   // (B,H,HD,T)
  uint16_t* sbf   = (uint16_t*)alloc((size_t)kB * kH * kT * kT * 2);
  // ff1 activations (kBT x kFF, 37.7MB) alias the dead q/k/vT/s region (53.5MB)
  uint16_t* ff1bf = qbf;

  // ---- weights fp32 -> bf16 (deterministic, every call) ----
  auto conv = [&](const float* s, uint16_t* dpt, size_t n) {
    f32_to_bf16_kernel<<<dim3((unsigned)((n + 255) / 256)), 256, 0, stream>>>(s, dpt, (int)n);
  };
  conv(wq,     wq_b,  (size_t)kL * kD * kD);
  conv(wk,     wk_b,  (size_t)kL * kD * kD);
  conv(wv,     wv_b,  (size_t)kL * kD * kD);
  conv(w_proj, wp_b,  (size_t)kL * kD * kD);
  conv(w_ff1,  wf1_b, (size_t)kL * kFF * kD);
  conv(w_ff2,  wf2_b, (size_t)kL * kD * kFF);
  conv(w_head, wh_b,  (size_t)kV * kD);

  // ---- embedding ----
  embed_kernel<<<dim3((kBT * kD) / 256), 256, 0, stream>>>(
      idx, condition, token_emb, w_cond, b_cond, pos_emb, x);

  const dim3 blk(256);
  const dim3 gD (kBT / 256, kD  / 64);   // (32, 9)
  const dim3 gFF(kBT / 256, kFF / 64);   // (32, 36)
  const dim3 gV (kBT / 256, kV  / 64);   // (32, 8)

  for (int l = 0; l < kL; ++l) {
    const uint16_t* wql = wq_b  + (size_t)l * kD * kD;
    const uint16_t* wkl = wk_b  + (size_t)l * kD * kD;
    const uint16_t* wvl = wv_b  + (size_t)l * kD * kD;
    const uint16_t* wpl = wp_b  + (size_t)l * kD * kD;
    const uint16_t* wf1 = wf1_b + (size_t)l * kFF * kD;
    const uint16_t* wf2 = wf2_b + (size_t)l * kD * kFF;

    // y = LN1(x)
    ln_bf16_kernel<<<kBT / 8, blk, 0, stream>>>(x, ln1_g + l * kD, ln1_b + l * kD, ybf);
    // q, k (bf16), v (stored transposed per head)
    gemm_wmma_kernel<1, false, false, false><<<gD, blk, 0, stream>>>(
        ybf, wql, nullptr, nullptr, qbf, kBT, kD, kD);
    gemm_wmma_kernel<1, false, false, false><<<gD, blk, 0, stream>>>(
        ybf, wkl, nullptr, nullptr, kbf, kBT, kD, kD);
    gemm_wmma_kernel<2, false, false, false><<<gD, blk, 0, stream>>>(
        ybf, wvl, nullptr, nullptr, vTbf, kBT, kD, kD);
    // S = QK^T * scale with causal mask
    attn_scores_kernel<<<dim3(kT / 32, kB * kH), blk, 0, stream>>>(qbf, kbf, sbf);
    // P = softmax(S) in place
    softmax_kernel<<<(kB * kH * kT) / 8, blk, 0, stream>>>(sbf);
    // att = P V
    attn_out_kernel<<<dim3(kHD / 32, kB * kH), blk, 0, stream>>>(sbf, vTbf, attbf);
    // x = x + att W_p^T + b_p
    gemm_wmma_kernel<0, false, true, true><<<gD, blk, 0, stream>>>(
        attbf, wpl, b_proj + l * kD, x, x, kBT, kD, kD);
    // y = LN2(x)
    ln_bf16_kernel<<<kBT / 8, blk, 0, stream>>>(x, ln2_g + l * kD, ln2_b + l * kD, ybf);
    // h = relu(y W1^T + b1)
    gemm_wmma_kernel<1, true, false, true><<<gFF, blk, 0, stream>>>(
        ybf, wf1, b_ff1 + (size_t)l * kFF, nullptr, ff1bf, kBT, kFF, kD);
    // x = x + h W2^T + b2
    gemm_wmma_kernel<0, false, true, true><<<gD, blk, 0, stream>>>(
        ff1bf, wf2, b_ff2 + l * kD, x, x, kBT, kD, kFF);
  }

  // final LN + head
  ln_bf16_kernel<<<kBT / 8, blk, 0, stream>>>(x, lnf_g, lnf_b, ybf);
  gemm_wmma_kernel<0, false, false, true><<<gV, blk, 0, stream>>>(
      ybf, wh_b, b_head, nullptr, (float*)d_out, kBT, kV, kD);
}
